// EmotionClassifier_90228672954461
// MI455X (gfx1250) — matrix operand order repaired and checked by reading, not compiled, not run on hardware
//
#include <hip/hip_runtime.h>

// =====================================================================
// EmotionClassifier (ResNet18 + 2x GATConv + FC) for gfx1250 (MI455X)
// - Convs: implicit-GEMM, v_wmma_f32_16x16x32_f16, f16 NHWC activations,
//   f32 accum, fused BN + residual + ReLU. Templated on conv geometry so
//   the K-loop is division-free; NB=2 cout-tiles per wave reuse each A
//   fragment across 2 WMMAs.
// - Weights packed on-device into the per-lane WMMA B-fragment layout.
// - GAT / pools / FC: small VALU kernels (latency-insignificant).
//
// Assumed d_in flattening (dict insertion order at top level, jax pytree
// alphabetical order inside 'params'):
//   0 image 1 landmarks 2 edge_index 3 fc_out.b 4 fc_out.w
//   5..8 gat1{att_dst,att_src,bias,w} 9..12 gat2{...}
//   13..16 resnet.bn1{b,g,m,v} 17 resnet.conv1
//   then layer1..4, per block: bn1{b,g,m,v}, bn2{b,g,m,v}, conv1, conv2
//   [, down_bn{b,g,m,v}, down_conv]
// =====================================================================

typedef _Float16 f16;
typedef __attribute__((ext_vector_type(16))) _Float16 v16h;
typedef __attribute__((ext_vector_type(8)))  _Float16 v8h;
typedef __attribute__((ext_vector_type(8)))  float    v8f;

struct ConvDescT {
  const f16* __restrict__ x;     // NHWC f16 input  [B,H,W,Cin]
  const f16* __restrict__ wpk;   // packed weights (WMMA B-fragment layout)
  const float* bnb; const float* bng; const float* bnm; const float* bnv;
  const f16* __restrict__ res;   // NHWC f16 residual or null
  f16* __restrict__ y;           // NHWC f16 output [B,OH,OW,Cout]
  int H, W, OH, OW, Cout, relu, Bn;
};

// One wave computes a 16(M=spatial) x (16*NB)(N=cout) tile.
// K order: k = (kh*KW+kw)*CIN + cin  (cin innermost == NHWC contiguous).
// A layout: row M = lane&15; lane<16 K{0-7,16-23}, lane>=16 K{8-15,24-31}.
template<int CIN, int KH, int KW, int STRIDE, int PAD, int NB>
__global__ __launch_bounds__(256)
void conv_wmma_t(ConvDescT d) {
  constexpr int KTOT = CIN * KH * KW;
  constexpr int NKC  = (KTOT + 31) / 32;

  const int lane = threadIdx.x & 31;
  const int wave = threadIdx.x >> 5;
  const int nTileN = d.Cout / (16 * NB);
  const int ohow   = d.OH * d.OW;
  const int nTileM = (d.Bn * ohow) >> 4;
  int tile = blockIdx.x * 8 + wave;                 // wave-uniform
  if (tile >= nTileM * nTileN) return;
  const int tn   = tile % nTileN;
  const int tm   = tile / nTileN;
  const int half = lane >> 4;

  // This lane's A-row spatial position
  const int m    = (tm << 4) + (lane & 15);
  const int b    = m / ohow;
  const int mrem = m - b * ohow;
  const int oh   = mrem / d.OW;
  const int ow   = mrem - oh * d.OW;

  v8f acc[NB] = {};
  const f16* __restrict__ wbase = d.wpk + (long)(tn * NB) * NKC * 512 + lane * 16;

  if constexpr ((CIN & 31) == 0) {
    // Fast path: every 32-wide K chunk stays inside one filter tap.
    const int ih0 = oh * STRIDE - PAD;
    const int iw0 = ow * STRIDE - PAD;
    const f16* xb = d.x + (long)b * d.H * d.W * CIN + 8 * half;
#pragma unroll
    for (int p = 0; p < KH * KW; ++p) {
      const int kh = p / KW, kw = p % KW;           // compile-time per unrolled p
      const int ih = ih0 + kh, iw = iw0 + kw;
      const bool inb = (ih >= 0) & (ih < d.H) & (iw >= 0) & (iw < d.W);
      const f16* px = xb + ((long)ih * d.W + iw) * CIN;
#pragma unroll 2
      for (int cc = 0; cc < CIN; cc += 32) {
        union { v16h v; f16 e[16]; } a;
        if (inb) {
          *(v8h*)&a.e[0] = *(const v8h*)(px + cc);
          *(v8h*)&a.e[8] = *(const v8h*)(px + cc + 16);
        } else {
#pragma unroll
          for (int i = 0; i < 16; ++i) a.e[i] = (f16)0.f;
        }
        const int kc = p * (CIN / 32) + (cc >> 5);
#pragma unroll
        for (int j = 0; j < NB; ++j) {
          const v16h bf = *(const v16h*)(wbase + (long)(j * NKC + kc) * 512);
          acc[j] = __builtin_amdgcn_wmma_f32_16x16x32_f16(
              false, a.v, false, bf, (short)0, acc[j], false, false);
        }
      }
    }
  } else {
    // Generic gather path (conv1: CIN=1, KTOT=49 zero-padded to 64).
#pragma unroll
    for (int kc = 0; kc < NKC; ++kc) {
      union { v16h v; f16 e[16]; } a;
#pragma unroll
      for (int u = 0; u < 2; ++u) {
        const int kb = kc * 32 + u * 16 + 8 * half;
#pragma unroll
        for (int i = 0; i < 8; ++i) {
          const int k = kb + i;
          f16 v = (f16)0.f;
          if (k < KTOT) {
            const int cin = k % CIN;
            const int p   = k / CIN;
            const int kw  = p % KW, kh = p / KW;
            const int ih  = oh * STRIDE - PAD + kh;
            const int iw  = ow * STRIDE - PAD + kw;
            if (ih >= 0 && ih < d.H && iw >= 0 && iw < d.W)
              v = d.x[(((long)b * d.H + ih) * d.W + iw) * CIN + cin];
          }
          a.e[u * 8 + i] = v;
        }
      }
#pragma unroll
      for (int j = 0; j < NB; ++j) {
        const v16h bf = *(const v16h*)(wbase + (long)(j * NKC + kc) * 512);
        acc[j] = __builtin_amdgcn_wmma_f32_16x16x32_f16(
            false, a.v, false, bf, (short)0, acc[j], false, false);
      }
    }
  }

  // Epilogue: C layout -> lane owns column n_j, VGPR r owns row r + 8*half.
  float sc[NB], bi[NB]; int nn[NB];
#pragma unroll
  for (int j = 0; j < NB; ++j) {
    const int n = (tn * NB + j) * 16 + (lane & 15);
    nn[j] = n;
    sc[j] = d.bng[n] * rsqrtf(d.bnv[n] + 1e-5f);
    bi[j] = d.bnb[n] - d.bnm[n] * sc[j];
  }
  int mr  = (tm << 4) + (half << 3);
  int bb  = mr / ohow;
  int rr  = mr - bb * ohow;
  int o_h = rr / d.OW;
  int o_w = rr - o_h * d.OW;
#pragma unroll
  for (int r = 0; r < 8; ++r) {
    const long rowbase = (((long)bb * d.OH + o_h) * d.OW + o_w) * d.Cout;
#pragma unroll
    for (int j = 0; j < NB; ++j) {
      float v = acc[j][r] * sc[j] + bi[j];
      if (d.res) v += (float)d.res[rowbase + nn[j]];
      if (d.relu) v = v > 0.f ? v : 0.f;
      d.y[rowbase + nn[j]] = (f16)v;
    }
    if (++o_w == d.OW) { o_w = 0; if (++o_h == d.OH) { o_h = 0; ++bb; } }
  }
}

// Pack f32 OIHW weights into per-lane WMMA B-fragment layout:
// packed[(nt*nK+kc)*512 + lane*16 + e] = W[n = nt*16+(lane&15)]
//   [k = kc*32 + (e>>3)*16 + (lane>>4)*8 + (e&7)], zero-padded past Ktot.
__global__ __launch_bounds__(256)
void pack_weights_kernel(const float* __restrict__ w, f16* __restrict__ o,
                         int Cout, int Cin, int KH, int KW) {
  int Ktot = Cin * KH * KW;
  int nK   = (Ktot + 31) >> 5;
  int total = Cout * nK * 32;          // total packed f16 elements
  int idx = blockIdx.x * 256 + threadIdx.x;
  if (idx >= total) return;
  int e    = idx & 15;
  int lane = (idx >> 4) & 31;
  int t2   = idx >> 9;                 // nt*nK + kc
  int kc   = t2 % nK;
  int nt   = t2 / nK;
  int n    = (nt << 4) + (lane & 15);
  int k    = kc * 32 + ((e >> 3) << 4) + ((lane >> 4) << 3) + (e & 7);
  f16 v = (f16)0.f;
  if (k < Ktot) {
    int cin = k % Cin; int p = k / Cin; int kw = p % KW; int kh = p / KW;
    v = (f16)w[(((long)n * Cin + cin) * KH + kh) * KW + kw];
  }
  o[idx] = v;
}

__global__ void img_to_f16_kernel(const float* __restrict__ x, f16* __restrict__ y, int n) {
  int i = blockIdx.x * 256 + threadIdx.x;
  if (i < n) y[i] = (f16)x[i];
}

// 3x3/s2/p1 maxpool: [32,112,112,64] -> [32,56,56,64], NHWC f16
__global__ void maxpool_kernel(const f16* __restrict__ x, f16* __restrict__ y) {
  int idx = blockIdx.x * 256 + threadIdx.x;
  const int total = 32 * 56 * 56 * 64;
  if (idx >= total) return;
  int c   = idx & 63;
  int pix = idx >> 6;
  int ow  = pix % 56;
  int oh  = (pix / 56) % 56;
  int b   = pix / (56 * 56);
  float mx = -1e30f;
  for (int kh = 0; kh < 3; ++kh)
    for (int kw = 0; kw < 3; ++kw) {
      int ih = oh * 2 - 1 + kh, iw = ow * 2 - 1 + kw;
      if (ih >= 0 && ih < 112 && iw >= 0 && iw < 112) {
        float v = (float)x[(((long)b * 112 + ih) * 112 + iw) * 64 + c];
        mx = v > mx ? v : mx;
      }
    }
  y[idx] = (f16)mx;
}

// [32,7,7,512] NHWC -> [32,512] f32 mean
__global__ void avgpool_kernel(const f16* __restrict__ x, float* __restrict__ f) {
  int t = blockIdx.x * 256 + threadIdx.x;
  if (t >= 32 * 512) return;
  int c = t & 511, b = t >> 9;
  float acc = 0.f;
  for (int p = 0; p < 49; ++p) acc += (float)x[((long)b * 49 + p) * 512 + c];
  f[t] = acc * (1.f / 49.f);
}

// h = landmarks @ W1 : [68,2]x[2,256], one block per (b,node)
__global__ void gat_lin1_kernel(const float* __restrict__ lm, const float* __restrict__ w,
                                float* __restrict__ h) {
  int c  = threadIdx.x;
  int bn = blockIdx.x;
  float x0 = lm[bn * 2 + 0], x1 = lm[bn * 2 + 1];
  h[(long)bn * 256 + c] = x0 * w[c] + x1 * w[256 + c];
}

// h = x @ W2 : [68,256]x[256,256], one block per (b,node), x row staged in LDS
__global__ void gat_lin256_kernel(const float* __restrict__ x, const float* __restrict__ w,
                                  float* __restrict__ h) {
  __shared__ float sx[256];
  int c  = threadIdx.x;
  int bn = blockIdx.x;
  sx[c] = x[(long)bn * 256 + c];
  __syncthreads();
  float acc = 0.f;
  for (int i = 0; i < 256; ++i) acc += sx[i] * w[(long)i * 256 + c];
  h[(long)bn * 256 + c] = acc;
}

// GATConv attention + aggregation + bias + ReLU, one block per sample.
// Edges = 256 from edge_index + 68 self loops (HEADS=4, DIM=64, N=68).
__global__ __launch_bounds__(256)
void gat_attn_kernel(const float* __restrict__ h,
                     const float* __restrict__ att_src, const float* __restrict__ att_dst,
                     const float* __restrict__ bias, const int* __restrict__ ei,
                     float* __restrict__ out) {
  const int b = blockIdx.x;
  const int tid = threadIdx.x;
  __shared__ int   ssrc[324], sdst[324];
  __shared__ float sas[272], sad[272], sm[272], sden[272];

  for (int e = tid; e < 324; e += 256) {
    if (e < 256) { ssrc[e] = ei[b * 512 + e]; sdst[e] = ei[b * 512 + 256 + e]; }
    else         { ssrc[e] = e - 256;         sdst[e] = e - 256; }
  }
  __syncthreads();

  for (int t = tid; t < 272; t += 256) {          // a_src/a_dst per (node, head)
    int node = t >> 2, hd = t & 3;
    const float* hr = h + ((long)b * 68 + node) * 256 + hd * 64;
    float as = 0.f, ad = 0.f;
    for (int dd = 0; dd < 64; ++dd) {
      as += hr[dd] * att_src[hd * 64 + dd];
      ad += hr[dd] * att_dst[hd * 64 + dd];
    }
    sas[t] = as; sad[t] = ad;
  }
  __syncthreads();

  for (int t = tid; t < 272; t += 256) {          // segment max + exp-sum
    int node = t >> 2, hd = t & 3;
    float adn = sad[t];
    float mx = -1e30f;
    for (int e = 0; e < 324; ++e)
      if (sdst[e] == node) {
        float v = sas[ssrc[e] * 4 + hd] + adn;
        v = v > 0.f ? v : 0.2f * v;
        mx = v > mx ? v : mx;
      }
    float den = 0.f;
    for (int e = 0; e < 324; ++e)
      if (sdst[e] == node) {
        float v = sas[ssrc[e] * 4 + hd] + adn;
        v = v > 0.f ? v : 0.2f * v;
        den += __expf(v - mx);
      }
    sm[t] = mx; sden[t] = den + 1e-16f;
  }
  __syncthreads();

  for (int t = tid; t < 68 * 256; t += 256) {     // weighted aggregation
    int node = t >> 8, c = t & 255, hd = c >> 6;
    float mx = sm[node * 4 + hd], den = sden[node * 4 + hd];
    float adn = sad[node * 4 + hd];
    float acc = 0.f;
    for (int e = 0; e < 324; ++e)
      if (sdst[e] == node) {
        int s = ssrc[e];
        float v = sas[s * 4 + hd] + adn;
        v = v > 0.f ? v : 0.2f * v;
        float wgt = __expf(v - mx) / den;
        acc += wgt * h[((long)b * 68 + s) * 256 + c];
      }
    float val = acc + bias[c];
    out[((long)b * 68 + node) * 256 + c] = val > 0.f ? val : 0.f;
  }
}

__global__ void gat_mean_kernel(const float* __restrict__ o, float* __restrict__ g) {
  int b = blockIdx.x, c = threadIdx.x;
  float acc = 0.f;
  for (int nde = 0; nde < 68; ++nde) acc += o[((long)b * 68 + nde) * 256 + c];
  g[b * 256 + c] = acc * (1.f / 68.f);
}

// out[b,cls] = concat(gat_out[b], feats[b]) @ Wfc + bfc
__global__ void fc_kernel(const float* __restrict__ g, const float* __restrict__ feats,
                          const float* __restrict__ w, const float* __restrict__ bias,
                          float* __restrict__ out) {
  int t = threadIdx.x;
  if (t >= 224) return;
  int b = t / 7, cls = t % 7;
  float acc = bias[cls];
  for (int i = 0; i < 256; ++i) acc += g[b * 256 + i] * w[i * 7 + cls];
  for (int j = 0; j < 512; ++j) acc += feats[b * 512 + j] * w[(256 + j) * 7 + cls];
  out[t] = acc;
}

extern "C" void kernel_launch(void* const* d_in, const int* in_sizes, int n_in,
                              void* d_out, int out_size, void* d_ws, size_t ws_size,
                              hipStream_t stream) {
  (void)in_sizes; (void)n_in; (void)out_size; (void)ws_size;
  constexpr int Bn = 32;

  const float* image     = (const float*)d_in[0];
  const float* landmarks = (const float*)d_in[1];
  const int*   edges     = (const int*)d_in[2];
  const float* fc_b      = (const float*)d_in[3];
  const float* fc_w      = (const float*)d_in[4];
  const float* g1_adst   = (const float*)d_in[5];
  const float* g1_asrc   = (const float*)d_in[6];
  const float* g1_bias   = (const float*)d_in[7];
  const float* g1_w      = (const float*)d_in[8];
  const float* g2_adst   = (const float*)d_in[9];
  const float* g2_asrc   = (const float*)d_in[10];
  const float* g2_bias   = (const float*)d_in[11];
  const float* g2_w      = (const float*)d_in[12];

  char* ws = (char*)d_ws;
  size_t off = 0;
  auto alloc = [&](size_t bytes) -> void* {
    off = (off + 255) & ~(size_t)255;
    void* p = ws + off;
    off += bytes;
    return p;
  };

  f16*   img16  = (f16*)alloc((size_t)Bn * 224 * 224 * 2);
  f16*   t0     = (f16*)alloc((size_t)Bn * 112 * 112 * 64 * 2);
  f16*   t1     = (f16*)alloc((size_t)Bn * 200704 * 2);
  f16*   t2     = (f16*)alloc((size_t)Bn * 200704 * 2);
  f16*   t3     = (f16*)alloc((size_t)Bn * 200704 * 2);
  float* h1     = (float*)alloc((size_t)Bn * 68 * 256 * 4);
  float* o1     = (float*)alloc((size_t)Bn * 68 * 256 * 4);
  float* h2     = (float*)alloc((size_t)Bn * 68 * 256 * 4);
  float* o2     = (float*)alloc((size_t)Bn * 68 * 256 * 4);
  float* feats  = (float*)alloc((size_t)Bn * 512 * 4);
  float* gatout = (float*)alloc((size_t)Bn * 256 * 4);

  { int n = Bn * 224 * 224;
    img_to_f16_kernel<<<(n + 255) / 256, 256, 0, stream>>>(image, img16, n); }

  // Pack weights + launch one conv (geometry as template args).
#define RUN_CONV(CIN, KH, KW, ST, PD, widx, bnbase, in_, out_, res_, H_, W_, COUT, RELU)   \
  do {                                                                                     \
    constexpr int nK_ = ((CIN) * (KH) * (KW) + 31) / 32;                                   \
    int ptotal_ = (COUT) * nK_ * 32;                                                       \
    f16* wpk_ = (f16*)alloc((size_t)ptotal_ * 2);                                          \
    pack_weights_kernel<<<(ptotal_ + 255) / 256, 256, 0, stream>>>(                        \
        (const float*)d_in[widx], wpk_, (COUT), (CIN), (KH), (KW));                        \
    int OH_ = ((H_) + 2 * (PD) - (KH)) / (ST) + 1;                                         \
    int OW_ = ((W_) + 2 * (PD) - (KW)) / (ST) + 1;                                         \
    ConvDescT dsc;                                                                         \
    dsc.x = (in_); dsc.wpk = wpk_;                                                         \
    dsc.bnb = (const float*)d_in[bnbase];     dsc.bng = (const float*)d_in[(bnbase) + 1];  \
    dsc.bnm = (const float*)d_in[(bnbase) + 2]; dsc.bnv = (const float*)d_in[(bnbase) + 3];\
    dsc.res = (res_); dsc.y = (out_);                                                      \
    dsc.H = (H_); dsc.W = (W_); dsc.OH = OH_; dsc.OW = OW_;                                \
    dsc.Cout = (COUT); dsc.relu = (RELU); dsc.Bn = Bn;                                     \
    int tiles_ = (Bn * OH_ * OW_ / 16) * ((COUT) / 32);                                    \
    conv_wmma_t<CIN, KH, KW, ST, PD, 2><<<(tiles_ + 7) / 8, 256, 0, stream>>>(dsc);        \
  } while (0)

  // stem
  RUN_CONV(1, 7, 7, 2, 3, 17, 13, img16, t0, nullptr, 224, 224, 64, 1);
  { int n = Bn * 56 * 56 * 64;
    maxpool_kernel<<<(n + 255) / 256, 256, 0, stream>>>(t0, t1); }
  // layer1
  RUN_CONV(64, 3, 3, 1, 1, 26, 18, t1, t2, nullptr, 56, 56, 64, 1);
  RUN_CONV(64, 3, 3, 1, 1, 27, 22, t2, t3, t1,      56, 56, 64, 1);
  RUN_CONV(64, 3, 3, 1, 1, 36, 28, t3, t2, nullptr, 56, 56, 64, 1);
  RUN_CONV(64, 3, 3, 1, 1, 37, 32, t2, t1, t3,      56, 56, 64, 1);
  // layer2
  RUN_CONV(64, 1, 1, 2, 0, 52, 48, t1, t3, nullptr, 56, 56, 128, 0);   // down
  RUN_CONV(64, 3, 3, 2, 1, 46, 38, t1, t2, nullptr, 56, 56, 128, 1);
  RUN_CONV(128, 3, 3, 1, 1, 47, 42, t2, t1, t3,     28, 28, 128, 1);
  RUN_CONV(128, 3, 3, 1, 1, 61, 53, t1, t2, nullptr, 28, 28, 128, 1);
  RUN_CONV(128, 3, 3, 1, 1, 62, 57, t2, t3, t1,      28, 28, 128, 1);
  // layer3
  RUN_CONV(128, 1, 1, 2, 0, 77, 73, t3, t1, nullptr, 28, 28, 256, 0);  // down
  RUN_CONV(128, 3, 3, 2, 1, 71, 63, t3, t2, nullptr, 28, 28, 256, 1);
  RUN_CONV(256, 3, 3, 1, 1, 72, 67, t2, t3, t1,      14, 14, 256, 1);
  RUN_CONV(256, 3, 3, 1, 1, 86, 78, t3, t2, nullptr, 14, 14, 256, 1);
  RUN_CONV(256, 3, 3, 1, 1, 87, 82, t2, t1, t3,      14, 14, 256, 1);
  // layer4
  RUN_CONV(256, 1, 1, 2, 0, 102, 98, t1, t3, nullptr, 14, 14, 512, 0); // down
  RUN_CONV(256, 3, 3, 2, 1, 96, 88, t1, t2, nullptr, 14, 14, 512, 1);
  RUN_CONV(512, 3, 3, 1, 1, 97, 92, t2, t1, t3,      7, 7, 512, 1);
  RUN_CONV(512, 3, 3, 1, 1, 111, 103, t1, t2, nullptr, 7, 7, 512, 1);
  RUN_CONV(512, 3, 3, 1, 1, 112, 107, t2, t3, t1,      7, 7, 512, 1);
#undef RUN_CONV

  avgpool_kernel<<<(32 * 512 + 255) / 256, 256, 0, stream>>>(t3, feats);

  // GAT branch
  gat_lin1_kernel<<<Bn * 68, 256, 0, stream>>>(landmarks, g1_w, h1);
  gat_attn_kernel<<<Bn, 256, 0, stream>>>(h1, g1_asrc, g1_adst, g1_bias, edges, o1);
  gat_lin256_kernel<<<Bn * 68, 256, 0, stream>>>(o1, g2_w, h2);
  gat_attn_kernel<<<Bn, 256, 0, stream>>>(h2, g2_asrc, g2_adst, g2_bias, edges, o2);
  gat_mean_kernel<<<Bn, 256, 0, stream>>>(o2, gatout);

  fc_kernel<<<1, 256, 0, stream>>>(gatout, feats, fc_w, fc_b, (float*)d_out);
}
